// SimpleRingAttentionWithHilbert_69776038690876
// MI455X (gfx1250) — compile-verified, matrix operand order checked by reading, not run on hardware
//
#include <hip/hip_runtime.h>
#include <cstdint>
#include <cstddef>

// ---------------------------------------------------------------------------
// Problem constants (B=1, S=4096, H=8, D=64; reference reshape mixes S and H:
// 8 virtual "heads", each = 4096 consecutive permuted (seq,head) rows).
// ---------------------------------------------------------------------------
#define SEQ   4096
#define HD    64
#define NHEAD 8
#define NTOK  (SEQ * NHEAD)          // 32768 rows of 64
#define QSCALE 0.18033688011112042f  // (1/sqrt(64)) * log2(e)

typedef _Float16 half_t;
typedef _Float16 v16h __attribute__((ext_vector_type(16)));
typedef float    v8f  __attribute__((ext_vector_type(8)));

__device__ __forceinline__ v8f wmma_f16(v16h a, v16h b, v8f c) {
  // v_wmma_f32_16x16x32_f16  D = A*B + C
  return __builtin_amdgcn_wmma_f32_16x16x32_f16(
      /*neg_a=*/false, a, /*neg_b=*/false, b,
      /*c_mod=*/(short)0, c, /*reuse_a=*/false, /*reuse_b=*/false);
}

// Load one 16x32 f16 A/B operand from a row-major f16 matrix.
// CDNA5 layout: lane L holds row (L&15); lanes 0-15 hold cols {0..7,16..23},
// lanes 16-31 hold cols {8..15,24..31}  ->  two 16-byte contiguous chunks.
__device__ __forceinline__ v16h load_ab16(const half_t* __restrict__ p, int rowStride) {
  const int lane = threadIdx.x & 31;
  const half_t* r = p + (size_t)(lane & 15) * rowStride + ((lane >> 4) << 3);
  uint4 lo = *reinterpret_cast<const uint4*>(r);        // cols c0 .. c0+7
  uint4 hi = *reinterpret_cast<const uint4*>(r + 16);   // cols c0+16 .. c0+23
  union { v16h h; uint32_t u[8]; } a;
  a.u[0] = lo.x; a.u[1] = lo.y; a.u[2] = lo.z; a.u[3] = lo.w;
  a.u[4] = hi.x; a.u[5] = hi.y; a.u[6] = hi.z; a.u[7] = hi.w;
  return a.h;
}

// ---------------------------------------------------------------------------
// Pack pass: gather with hilbert perm, fp32 -> f16.
//   Qp[g][d] = q[perm[g>>3]*8 + (g&7)][d] * QSCALE   (g = head*4096 + m)
//   Kp[g][d] = k[...]
//   VT[head][d][m] = v[...]          (V transposed so PV B.. loads row-major)
// ---------------------------------------------------------------------------
__global__ __launch_bounds__(256) void hilbert_pack_kernel(
    const float* __restrict__ q, const float* __restrict__ k,
    const float* __restrict__ v, const int* __restrict__ perm,
    half_t* __restrict__ Qp, half_t* __restrict__ Kp, half_t* __restrict__ VT) {
  const int e = blockIdx.x * 256 + threadIdx.x;   // e in [0, NTOK*HD)

  // Q/K: row-major destination
  {
    const int g = e >> 6, d = e & 63;
    const int src = (perm[g >> 3] * 8 + (g & 7)) * 64 + d;
    Qp[e] = (half_t)(q[src] * QSCALE);
    Kp[e] = (half_t)(k[src]);
  }
  // V: transposed destination VT[head][dim][m], write-contiguous in e
  {
    const int i  = e >> 18;            // head (64*4096 = 2^18 per head)
    const int re = e & ((1 << 18) - 1);
    const int d2 = re >> 12;           // dim
    const int m2 = re & 4095;          // key index within head
    const int g2 = (i << 12) | m2;
    const int src2 = (perm[g2 >> 3] * 8 + (g2 & 7)) * 64 + d2;
    VT[e] = (half_t)(v[src2]);
  }
}

// ---------------------------------------------------------------------------
// Flash attention, transposed-score formulation.
// Block = 8 waves x 32 = 256 threads; wave owns 16 queries, full D=64.
// Per 32-key step: S^T tiles via 4 WMMAs, per-lane softmax (+1 xor-16
// shuffle), P^T (C-layout) feeds PV WMMAs directly as the B operand.
// ---------------------------------------------------------------------------
__global__ __launch_bounds__(256) void hilbert_attn_kernel(
    const half_t* __restrict__ Qp, const half_t* __restrict__ Kp,
    const half_t* __restrict__ VT, const int* __restrict__ perm,
    float* __restrict__ out) {
  __shared__ float lds[8][16][64];    // [wave][query][dim]

  const int lane = threadIdx.x & 31;
  const int wave = threadIdx.x >> 5;
  const int head = blockIdx.y;
  const int mBase = blockIdx.x * 128 + wave * 16;       // query tile base

  const half_t* __restrict__ Qh = Qp + ((size_t)head * SEQ + mBase) * HD;
  const half_t* __restrict__ Kh = Kp + (size_t)head * SEQ * HD;
  const half_t* __restrict__ Vh = VT + (size_t)head * HD * SEQ;

  // Q operand (B-matrix of S^T = K * Q^T), both 32-wide d-chunks, persistent.
  const v16h qb0 = load_ab16(Qh +  0, HD);
  const v16h qb1 = load_ab16(Qh + 32, HD);

  v8f acc0 = {}, acc1 = {}, acc2 = {}, acc3 = {};  // O^T: dim tiles 0..3 x 16 queries
  float m_run = -INFINITY, l_run = 0.0f;

  for (int kb = 0; kb < SEQ; kb += 32) {
    const half_t* Kt0 = Kh + (size_t)kb * HD;
    const half_t* Kt1 = Kh + (size_t)(kb + 16) * HD;
    const v16h ka00 = load_ab16(Kt0, HD);        // keys kb..+15 , d 0..31
    const v16h ka01 = load_ab16(Kt0 + 32, HD);   // keys kb..+15 , d 32..63
    const v16h ka10 = load_ab16(Kt1, HD);
    const v16h ka11 = load_ab16(Kt1 + 32, HD);

    v8f st0 = {}; v8f st1 = {};
    st0 = wmma_f16(ka00, qb0, st0);
    st0 = wmma_f16(ka01, qb1, st0);              // S^T keys kb..kb+15
    st1 = wmma_f16(ka10, qb0, st1);
    st1 = wmma_f16(ka11, qb1, st1);              // S^T keys kb+16..kb+31

    // --- online softmax (log2 domain; scale folded into Qp) -------------
    float mloc = st0[0];
#pragma unroll
    for (int j = 1; j < 8; ++j) mloc = fmaxf(mloc, st0[j]);
#pragma unroll
    for (int j = 0; j < 8; ++j) mloc = fmaxf(mloc, st1[j]);
    mloc = fmaxf(mloc, __shfl_xor(mloc, 16, 32));     // join lane halves
    const float m_new = fmaxf(m_run, mloc);
    const float corr  = __builtin_amdgcn_exp2f(m_run - m_new);

    v16h pb;            // P^T in B-operand layout: all in-lane, no shuffles
    float lsum = 0.0f;
#pragma unroll
    for (int j = 0; j < 8; ++j) {
      const float p = __builtin_amdgcn_exp2f(st0[j] - m_new);
      lsum += p;
      pb[j] = (half_t)p;
    }
#pragma unroll
    for (int j = 0; j < 8; ++j) {
      const float p = __builtin_amdgcn_exp2f(st1[j] - m_new);
      lsum += p;
      pb[8 + j] = (half_t)p;
    }
    lsum += __shfl_xor(lsum, 16, 32);
    l_run = l_run * corr + lsum;
    m_run = m_new;

    acc0 *= corr; acc1 *= corr; acc2 *= corr; acc3 *= corr;

    // --- O^T += V^T * P^T -----------------------------------------------
    const v16h va0 = load_ab16(Vh + (size_t)0  * SEQ + kb, SEQ);
    const v16h va1 = load_ab16(Vh + (size_t)16 * SEQ + kb, SEQ);
    const v16h va2 = load_ab16(Vh + (size_t)32 * SEQ + kb, SEQ);
    const v16h va3 = load_ab16(Vh + (size_t)48 * SEQ + kb, SEQ);
    acc0 = wmma_f16(va0, pb, acc0);
    acc1 = wmma_f16(va1, pb, acc1);
    acc2 = wmma_f16(va2, pb, acc2);
    acc3 = wmma_f16(va3, pb, acc3);
  }

  const float inv = 1.0f / l_run;
  acc0 *= inv; acc1 *= inv; acc2 *= inv; acc3 *= inv;

  // Epilogue: O^T (C layout) -> LDS -> coalesced row stores with perm scatter.
  {
    const int qcol = lane & 15;
    const int roff = (lane >> 4) * 8;
#pragma unroll
    for (int j = 0; j < 8; ++j) {
      lds[wave][qcol][ 0 + roff + j] = acc0[j];
      lds[wave][qcol][16 + roff + j] = acc1[j];
      lds[wave][qcol][32 + roff + j] = acc2[j];
      lds[wave][qcol][48 + roff + j] = acc3[j];
    }
  }
  __syncthreads();
#pragma unroll
  for (int r = 0; r < 16; ++r) {
    const int srow = perm[mBase + r];                 // out[s=perm[m], h=head]
    const float2 val = *reinterpret_cast<const float2*>(&lds[wave][r][lane * 2]);
    *reinterpret_cast<float2*>(out + ((size_t)srow * NHEAD + head) * HD + lane * 2) = val;
  }
}

// ---------------------------------------------------------------------------
extern "C" void kernel_launch(void* const* d_in, const int* in_sizes, int n_in,
                              void* d_out, int out_size, void* d_ws, size_t ws_size,
                              hipStream_t stream) {
  const float* q    = (const float*)d_in[0];
  const float* k    = (const float*)d_in[1];
  const float* v    = (const float*)d_in[2];
  const int*   perm = (const int*)d_in[3];
  float*       out  = (float*)d_out;

  half_t* Qp = (half_t*)d_ws;                        // 4 MiB
  half_t* Kp = Qp + (size_t)NTOK * HD;               // 4 MiB
  half_t* VT = Kp + (size_t)NTOK * HD;               // 4 MiB  (total 12 MiB)

  hilbert_pack_kernel<<<(NTOK * HD) / 256, 256, 0, stream>>>(q, k, v, perm, Qp, Kp, VT);

  dim3 grid(SEQ / 128, NHEAD);                       // 32 x 8 blocks, 8 waves each
  hilbert_attn_kernel<<<grid, 256, 0, stream>>>(Qp, Kp, VT, perm, out);
}
